// SoftCodebookGate_61701500175228
// MI455X (gfx1250) — compile-verified, matrix operand order checked by reading, not run on hardware
//
#include <hip/hip_runtime.h>

#define DIN   1024
#define TILE_M  16
#define TAU_F 10.0f

typedef __attribute__((ext_vector_type(16))) __bf16 bf16x16;
typedef __attribute__((ext_vector_type(8)))  float  floatx8;

union FragU { bf16x16 v; uint4 q[2]; };

__device__ __forceinline__ unsigned short f32_to_bf16(float f) {
  unsigned int x = __float_as_uint(f);
  unsigned int r = x + 0x7FFFu + ((x >> 16) & 1u);   // round-to-nearest-even
  return (unsigned short)(r >> 16);
}

__device__ __forceinline__ floatx8 wmma_bf16(const FragU& a, const FragU& b, floatx8 c) {
  return __builtin_amdgcn_wmma_f32_16x16x32_bf16(false, a.v, false, b.v,
                                                 (short)0, c, false, false);
}

// ---------------------------------------------------------------------------
// K1: per-row L2 normalize (d = 1024) + convert to bf16.  grid = nrows, block = 256.
// ---------------------------------------------------------------------------
__global__ void rownorm_bf16_kernel(const float* __restrict__ src,
                                    unsigned short* __restrict__ dst) {
  __shared__ float wsum[8];
  const int t    = threadIdx.x;          // 256 threads, 4 floats each = 1024
  const int lane = t & 31, wave = t >> 5;
  const size_t row = blockIdx.x;

  const float4 v = ((const float4*)(src + row * DIN))[t];
  float ss = v.x * v.x + v.y * v.y + v.z * v.z + v.w * v.w;
  #pragma unroll
  for (int off = 16; off; off >>= 1) ss += __shfl_xor(ss, off, 32);
  if (lane == 0) wsum[wave] = ss;
  __syncthreads();
  float tot = 0.f;
  #pragma unroll
  for (int i = 0; i < 8; i++) tot += wsum[i];
  const float inv = rsqrtf(tot + 1e-12f);

  ushort4 o;
  o.x = f32_to_bf16(v.x * inv);
  o.y = f32_to_bf16(v.y * inv);
  o.z = f32_to_bf16(v.z * inv);
  o.w = f32_to_bf16(v.w * inv);
  ((ushort4*)(dst + row * DIN))[t] = o;
}

// ---------------------------------------------------------------------------
// K2: fused bf16-WMMA GEMM (16 rows x K cols, reduce over 1024) + top-8 + softmax.
// grid = M/16, block = 256 (8 waves). Dynamic LDS: 16x1024 bf16 A-tile + 16xK f32 logits.
// ---------------------------------------------------------------------------
__global__ void gemm_topk_kernel(const unsigned short* __restrict__ zfb,
                                 const unsigned short* __restrict__ cfb,
                                 float* __restrict__ alphaOut,
                                 int*   __restrict__ idxOut,
                                 int K) {
  extern __shared__ __align__(16) char smem[];
  unsigned short* ldsA = (unsigned short*)smem;                        // 32 KB @ LDS off 0
  float* ldsL = (float*)(smem + (size_t)TILE_M * DIN * sizeof(unsigned short)); // 16*K f32

  const int tid  = threadIdx.x;
  const int lane = tid & 31;
  const int wave = tid >> 5;
  const size_t rowBase = (size_t)blockIdx.x * TILE_M;

  // ---- stage A tile into LDS via CDNA5 async global->LDS DMA (ASYNCcnt) ----
  {
    const char* srcA = (const char*)(zfb + rowBase * DIN);   // 32 KB, 2048 x 16B
    const int totalBytes = TILE_M * DIN * 2;
    for (int i = tid * 16; i < totalBytes; i += 256 * 16) {
      const unsigned int ldsOff = (unsigned int)i;           // smem base == LDS offset 0
      const char* g = srcA + i;
      asm volatile("global_load_async_to_lds_b128 %0, %1, off"
                   :: "v"(ldsOff), "v"(g) : "memory");
    }
    asm volatile("s_wait_asynccnt 0x0" ::: "memory");
  }
  __syncthreads();

  // 16-bit A operand lane layout: lanes 0-15 hold K+{0..7,16..23}, lanes 16-31 hold K+{8..15,24..31}
  const int koff = (lane >= 16) ? 8 : 0;
  const unsigned short* aBase = ldsA + (size_t)(lane & 15) * DIN + koff;

  // Each wave computes two 16x16 N-tiles per pass (shared A fragment),
  // with fragment loads software-pipelined one k-step ahead.
  const int nPairs = K >> 5;                     // pairs of 16-col tiles
  for (int pair = wave; pair < nPairs; pair += 8) {
    const int nbase = pair << 5;
    const unsigned short* b0 = cfb + (size_t)(nbase + (lane & 15)) * DIN + koff;
    const unsigned short* b1 = b0 + (size_t)16 * DIN;

    floatx8 acc0 = {}, acc1 = {};
    FragU fa, fb0, fb1;
    fa.q[0]  = *(const uint4*)(aBase);      fa.q[1]  = *(const uint4*)(aBase + 16);
    fb0.q[0] = *(const uint4*)(b0);         fb0.q[1] = *(const uint4*)(b0 + 16);
    fb1.q[0] = *(const uint4*)(b1);         fb1.q[1] = *(const uint4*)(b1 + 16);

    #pragma unroll 2
    for (int kb = 0; kb < DIN; kb += 32) {
      const int kn = (kb + 32) & (DIN - 1);        // wraps to 0 on last step (harmless)
      FragU na, nb0, nb1;
      na.q[0]  = *(const uint4*)(aBase + kn);  na.q[1]  = *(const uint4*)(aBase + kn + 16);
      nb0.q[0] = *(const uint4*)(b0 + kn);     nb0.q[1] = *(const uint4*)(b0 + kn + 16);
      nb1.q[0] = *(const uint4*)(b1 + kn);     nb1.q[1] = *(const uint4*)(b1 + kn + 16);

      acc0 = wmma_bf16(fa, fb0, acc0);
      acc1 = wmma_bf16(fa, fb1, acc1);

      fa = na; fb0 = nb0; fb1 = nb1;
    }

    // C/D layout: element (m, n): m = v + 8*(lane>=16), n = lane&15
    const int mhi  = (lane >> 4) << 3;
    const int col0 = nbase + (lane & 15);
    #pragma unroll
    for (int v = 0; v < 8; v++) {
      ldsL[(size_t)(mhi + v) * K + col0]      = TAU_F * acc0[v];
      ldsL[(size_t)(mhi + v) * K + col0 + 16] = TAU_F * acc1[v];
    }
  }
  __syncthreads();

  // ---- top-8 + softmax: wave w owns rows 2w, 2w+1 ----
  for (int rr = 0; rr < 2; rr++) {
    const int r = wave * 2 + rr;
    float* rowp = ldsL + (size_t)r * K;
    float vals[8]; int ids[8];
    #pragma unroll 1
    for (int i = 0; i < 8; i++) {
      float best = -1e30f; int bidx = 0x7FFFFFFF;
      for (int j = lane; j < K; j += 32) {
        float x = rowp[j];
        if (x > best) { best = x; bidx = j; }
      }
      #pragma unroll
      for (int off = 16; off; off >>= 1) {
        float ov = __shfl_xor(best, off, 32);
        int   oi = __shfl_xor(bidx, off, 32);
        if (ov > best || (ov == best && oi < bidx)) { best = ov; bidx = oi; }
      }
      vals[i] = best; ids[i] = bidx;
      if (lane == 0) rowp[bidx] = -1e30f;             // knock out current max
      asm volatile("s_wait_dscnt 0x0" ::: "memory");  // make LDS write visible wave-wide
    }
    const float mx = vals[0];
    float ex[8]; float sum = 0.f;
    #pragma unroll
    for (int i = 0; i < 8; i++) { ex[i] = __expf(vals[i] - mx); sum += ex[i]; }
    const float inv = 1.0f / sum;
    if (lane == 0) {
      const size_t orow = (rowBase + (size_t)r) * 8;
      #pragma unroll
      for (int i = 0; i < 8; i++) {
        alphaOut[orow + i] = ex[i] * inv;
        idxOut[orow + i]   = ids[i];
      }
    }
  }
}

// ---------------------------------------------------------------------------
// K3: out = target * (1 + sum_i alpha_i * E[idx_i]).  grid = M, block = 256 (float4).
// ---------------------------------------------------------------------------
__global__ void gather_gate_kernel(const float* __restrict__ target,
                                   const float* __restrict__ E,
                                   const float* __restrict__ alpha,
                                   const int*   __restrict__ idx,
                                   float* __restrict__ out) {
  const size_t row = blockIdx.x;
  const int t = threadIdx.x;   // 256 threads x float4 = 1024 cols
  float a[8]; int id[8];
  #pragma unroll
  for (int i = 0; i < 8; i++) { a[i] = alpha[row * 8 + i]; id[i] = idx[row * 8 + i]; }

  float4 g = {0.f, 0.f, 0.f, 0.f};
  #pragma unroll
  for (int i = 0; i < 8; i++) {
    const float4 e = ((const float4*)(E + (size_t)id[i] * DIN))[t];
    g.x += a[i] * e.x; g.y += a[i] * e.y; g.z += a[i] * e.z; g.w += a[i] * e.w;
  }
  const float4 tg = ((const float4*)(target + row * DIN))[t];
  float4 r;
  r.x = tg.x * (1.f + g.x);
  r.y = tg.y * (1.f + g.y);
  r.z = tg.z * (1.f + g.z);
  r.w = tg.w * (1.f + g.w);
  ((float4*)(out + row * DIN))[t] = r;
}

// ---------------------------------------------------------------------------
extern "C" void kernel_launch(void* const* d_in, const int* in_sizes, int n_in,
                              void* d_out, int out_size, void* d_ws, size_t ws_size,
                              hipStream_t stream) {
  const float* z        = (const float*)d_in[0];  // [M, 1024]
  const float* target   = (const float*)d_in[1];  // [M, 1024]
  const float* codebook = (const float*)d_in[2];  // [K, 1024]
  const float* E        = (const float*)d_in[3];  // [K, 1024]
  float* out = (float*)d_out;

  const int M = in_sizes[0] / DIN;   // 16384
  const int K = in_sizes[2] / DIN;   // 4096

  // workspace carve (all chunks 256B-aligned by construction)
  char* ws = (char*)d_ws;
  unsigned short* zfb = (unsigned short*)ws;                 ws += (size_t)M * DIN * sizeof(unsigned short);
  unsigned short* cfb = (unsigned short*)ws;                 ws += (size_t)K * DIN * sizeof(unsigned short);
  float* alpha        = (float*)ws;                          ws += (size_t)M * 8 * sizeof(float);
  int*   idx          = (int*)ws;

  // K1: normalize + bf16
  rownorm_bf16_kernel<<<M, 256, 0, stream>>>(z, zfb);
  rownorm_bf16_kernel<<<K, 256, 0, stream>>>(codebook, cfb);

  // K2: fused WMMA GEMM + top-8 + softmax
  const size_t smemBytes = (size_t)TILE_M * DIN * sizeof(unsigned short)
                         + (size_t)TILE_M * K * sizeof(float);     // 288 KB < 320 KB/WGP
  hipFuncSetAttribute(reinterpret_cast<const void*>(gemm_topk_kernel),
                      hipFuncAttributeMaxDynamicSharedMemorySize, (int)smemBytes);
  gemm_topk_kernel<<<M / TILE_M, 256, smemBytes, stream>>>(zfb, cfb, alpha, idx, K);

  // K3: gather + gate (bandwidth-bound; E is L2-resident)
  gather_gate_kernel<<<M, 256, 0, stream>>>(target, E, alpha, idx, out);
}
